// PatientDeepPoolEncoder_16612933501059
// MI455X (gfx1250) — matrix operand order repaired and checked by reading, not compiled, not run on hardware
//
#include <hip/hip_runtime.h>

// ---------------- problem constants ----------------
#define NPTS 32
#define LSEQ 2048
#define CTAB 256
#define DEMI 8
#define DEMH 40
#define DEMO 20
#define KTOT 1044           // 4*256 + 20
#define KSTEPS 33           // 33*32 = 1056 (zero padded K)
#define OUTW 788            // 3*256 + 20
#define NCHUNK 16           // L split into 16 chunks of 128 for the scan
#define CHUNK 128

typedef __attribute__((ext_vector_type(16))) __bf16 v16bf;
typedef __attribute__((ext_vector_type(2)))  __bf16 v2bf;
typedef __attribute__((ext_vector_type(8)))  float  v8f;
typedef __attribute__((ext_vector_type(16))) float  v16f;

union BF16Frag { unsigned int u[8]; v16bf v; };

// f32x2 -> packed bf16x2 (RNE) via hardware packed convert (staging path)
static __device__ __forceinline__ unsigned int pack_bf16(float a, float b) {
#if __has_builtin(__builtin_amdgcn_cvt_pk_bf16_f32)
  union { v2bf v; unsigned int u; } cv;
  cv.v = __builtin_amdgcn_cvt_pk_bf16_f32(a, b);
  return cv.u;
#else
  union { __bf16 h[2]; unsigned int u; } cv;
  cv.h[0] = (__bf16)a;
  cv.h[1] = (__bf16)b;
  return cv.u;
#endif
}

// ---------------- kernel 1: dem MLP 8->40->20 (tiny) ----------------
__global__ void dem_mlp_kernel(const float* __restrict__ dem,
                               const float* __restrict__ W1, const float* __restrict__ b1,
                               const float* __restrict__ W2, const float* __restrict__ b2,
                               float* __restrict__ dvec) {
  int i = threadIdx.x;
  if (i >= NPTS) return;
  float x[DEMI];
  for (int k = 0; k < DEMI; ++k) x[k] = dem[i * DEMI + k];
  float h[DEMH];
  for (int j = 0; j < DEMH; ++j) {
    float s = b1[j];
    for (int k = 0; k < DEMI; ++k) s += x[k] * W1[k * DEMH + j];
    h[j] = fmaxf(s, 0.f);
  }
  for (int j = 0; j < DEMO; ++j) {
    float s = b2[j];
    for (int k = 0; k < DEMH; ++k) s += h[k] * W2[k * DEMO + j];
    dvec[i * DEMO + j] = fmaxf(s, 0.f);
  }
}

// ---------------- kernel 2: big GEMM via bf16 WMMA ----------------
// x[r, c] = sum_k concat(ts0..ts3, dem)[r, k] * W_lin[k, c] + b_lin[c]
// grid: (16 N-tiles, 512 M-tiles), block 256 = 8 wave32, wave -> 16x16 tile.
__global__ __launch_bounds__(256)
void gemm_kernel(const float* __restrict__ ts0, const float* __restrict__ ts1,
                 const float* __restrict__ ts2, const float* __restrict__ ts3,
                 const float* __restrict__ dvec, const float* __restrict__ Wlin,
                 const float* __restrict__ blin, float* __restrict__ xbuf) {
  // B fragments staged in LDS, fragment-order, pad stride 9 dwords (bank-conflict free)
  __shared__ unsigned int Bs[KSTEPS][32][9];
  const int nb = blockIdx.x;   // N tile (16 cols)
  const int mt = blockIdx.y;   // M tile (128 rows)
  const int t  = threadIdx.x;

  // Stage W_lin[:, nb*16 .. +16) as bf16 pairs in exact B-matrix lane layout:
  // lane = (khalf<<4)|nloc ; dword i holds K = ks*32 + khalf*8 + 2i + (i>=4 ? 8 : 0) (,+1)
  for (int idx = t; idx < KSTEPS * 32 * 8; idx += 256) {
    int ks   = idx >> 8;         // /256
    int rem  = idx & 255;
    int lane = rem >> 3;
    int i    = rem & 7;
    int nloc = lane & 15;
    int k    = ks * 32 + ((lane >> 4) << 3) + 2 * i + ((i >> 2) << 3);
    int gcol = nb * 16 + nloc;
    float v0 = (k     < KTOT) ? Wlin[(size_t)k       * CTAB + gcol] : 0.f;
    float v1 = (k + 1 < KTOT) ? Wlin[(size_t)(k + 1) * CTAB + gcol] : 0.f;
    Bs[ks][lane][i] = pack_bf16(v0, v1);
  }
  __syncthreads();

  const int wave = t >> 5;
  const int lane = t & 31;
  const int m    = lane & 15;   // A row within tile for this lane
  const int half = lane >> 4;
  const int arow = mt * 128 + wave * 16 + m;   // global row (n*L + l) for A loads
  const int nrow = arow >> 11;                 // patient index of this row

  v8f c = {};

  // 32 K-steps covering ts0..ts3; tab unrolled so the table pointer is static
#pragma unroll
  for (int tab = 0; tab < 4; ++tab) {
    const float* tp = (tab == 0) ? ts0 : (tab == 1) ? ts1 : (tab == 2) ? ts2 : ts3;
    const float* rowp = tp + (size_t)arow * CTAB;
#pragma unroll 2
    for (int kss = 0; kss < 8; ++kss) {
      int ks = tab * 8 + kss;
      // lane's 16 K-values = two contiguous 8-float runs -> 2x v8f (4x b128) loads
      const float* base = rowp + kss * 32 + half * 8;
      v8f r0 = *(const v8f*)(base);        // K = kk+0..7
      v8f r1 = *(const v8f*)(base + 16);   // K = 16+kk+0..7
      v16f af = __builtin_shufflevector(r0, r1, 0, 1, 2, 3, 4, 5, 6, 7,
                                        8, 9, 10, 11, 12, 13, 14, 15);
      v16bf av = __builtin_convertvector(af, v16bf);   // 8x v_cvt_pk_bf16_f32
      BF16Frag b;
#pragma unroll
      for (int i = 0; i < 8; ++i) b.u[i] = Bs[ks][lane][i];
      c = __builtin_amdgcn_wmma_f32_16x16x32_bf16(false, av, false, b.v,
                                                  (short)0, c, false, false);
    }
  }
  // tail K-step: K = 1024..1055 -> dem features (20 valid, rest zero)
  {
    v16f af;
#pragma unroll
    for (int e = 0; e < 16; ++e) {
      int koff = ((e >> 3) ? 16 : 0) + half * 8 + (e & 7);  // runs mirror main loop
      af[e] = (koff < DEMO) ? dvec[nrow * DEMO + koff] : 0.f;
    }
    v16bf av = __builtin_convertvector(af, v16bf);
    BF16Frag b;
#pragma unroll
    for (int i = 0; i < 8; ++i) b.u[i] = Bs[32][lane][i];
    c = __builtin_amdgcn_wmma_f32_16x16x32_bf16(false, av, false, b.v,
                                                (short)0, c, false, false);
  }

  // C/D layout: vgpr i -> M = i + 8*half, N = lane&15
  const int col  = nb * 16 + (lane & 15);
  const float bv = blin[col];
#pragma unroll
  for (int i = 0; i < 8; ++i) {
    int mrow = mt * 128 + wave * 16 + i + half * 8;
    xbuf[(size_t)mrow * CTAB + col] = c[i] + bv;
  }
}

// ---------------- kernel 3a: per-chunk scan aggregates ----------------
__global__ __launch_bounds__(256)
void scan_agg_kernel(const float* __restrict__ xbuf,
                     float* __restrict__ aggmax, int* __restrict__ aggarg,
                     float* __restrict__ aggsum) {
  int blk = blockIdx.x;          // n*NCHUNK + ch
  int n = blk >> 4, ch = blk & (NCHUNK - 1);
  int c = threadIdx.x;
  float mval = -__builtin_inff(); int marg = 0; float s = 0.f;
  size_t base = ((size_t)n * LSEQ + (size_t)ch * CHUNK) * CTAB + c;
  for (int i = 0; i < CHUNK; ++i) {
    float x = xbuf[base + (size_t)i * CTAB];
    if (x > mval) { mval = x; marg = ch * CHUNK + i; }
    s += x;
  }
  int o = blk * CTAB + c;
  aggmax[o] = mval; aggarg[o] = marg; aggsum[o] = s;
}

// ---------------- kernel 3b: exclusive carries across chunks ----------------
__global__ __launch_bounds__(256)
void scan_carry_kernel(const float* __restrict__ aggmax, const int* __restrict__ aggarg,
                       const float* __restrict__ aggsum,
                       float* __restrict__ carrymax, int* __restrict__ carryarg,
                       float* __restrict__ carrysum) {
  int n = blockIdx.x;            // 32
  int c = threadIdx.x;           // 256
  float cm = -__builtin_inff(); int ca = 0; float cs = 0.f;
  for (int ch = 0; ch < NCHUNK; ++ch) {
    int o = (n * NCHUNK + ch) * CTAB + c;
    carrymax[o] = cm; carryarg[o] = ca; carrysum[o] = cs;
    float am = aggmax[o];
    if (am > cm) { cm = am; ca = aggarg[o]; }
    cs += aggsum[o];
  }
}

// ---------------- kernel 3c: final scan + fused LayerNorm + outputs ----------------
// One block = one (n, chunk): 256 threads (one per channel), 128 sequential l's.
__global__ __launch_bounds__(256)
void scan_final_kernel(const float* __restrict__ xbuf,
                       const float* __restrict__ carrymax, const int* __restrict__ carryarg,
                       const float* __restrict__ carrysum, const float* __restrict__ dvec,
                       float* __restrict__ out, int* __restrict__ tinds,
                       float* __restrict__ acts) {
  __shared__ float red[8][6];
  int blk = blockIdx.x;
  int n = blk >> 4, ch = blk & (NCHUNK - 1);
  int c = threadIdx.x;
  int wid = c >> 5, lane = c & 31;
  int o = blk * CTAB + c;
  float mval = carrymax[o]; int marg = carryarg[o]; float csum = carrysum[o];
  float dval = (c < DEMO) ? dvec[n * DEMO + c] : 0.f;
  int tb4[4]; float ab4[4];

  for (int i = 0; i < CHUNK; ++i) {
    int l = ch * CHUNK + i;
    float x = xbuf[((size_t)n * LSEQ + l) * CTAB + c];
    if (x > mval) { mval = x; marg = l; }
    csum += x;
    bool pad = (l < LSEQ - 1);
    float pm = pad ? fmaxf(mval, 0.f) : mval;                 // pad zero in window
    int ti = (pad && mval <= 0.f) ? (l - (LSEQ - 1)) : marg;  // torch padded-argmax rule
    float pa = csum * (1.0f / LSEQ);
    float ps = csum * rsqrtf((float)(l + 1));

    // transposed outputs (pre-LayerNorm p_max), buffered into 16B stores
    tb4[i & 3] = ti; ab4[i & 3] = pm;
    if ((i & 3) == 3) {
      size_t tb = ((size_t)n * CTAB + c) * LSEQ + (l - 3);
      *(int4*)(tinds + tb)  = make_int4(tb4[0], tb4[1], tb4[2], tb4[3]);
      *(float4*)(acts + tb) = make_float4(ab4[0], ab4[1], ab4[2], ab4[3]);
    }

    // fused affine-free LayerNorm over the 256 channels for pm/pa/ps
    float s0 = pm, q0 = pm * pm, s1 = pa, q1 = pa * pa, s2 = ps, q2 = ps * ps;
#pragma unroll
    for (int off = 16; off > 0; off >>= 1) {
      s0 += __shfl_xor(s0, off, 32); q0 += __shfl_xor(q0, off, 32);
      s1 += __shfl_xor(s1, off, 32); q1 += __shfl_xor(q1, off, 32);
      s2 += __shfl_xor(s2, off, 32); q2 += __shfl_xor(q2, off, 32);
    }
    __syncthreads();  // protect red[] from previous iteration
    if (lane == 0) {
      red[wid][0] = s0; red[wid][1] = q0; red[wid][2] = s1;
      red[wid][3] = q1; red[wid][4] = s2; red[wid][5] = q2;
    }
    __syncthreads();
    float S0 = 0, Q0 = 0, S1 = 0, Q1 = 0, S2 = 0, Q2 = 0;
#pragma unroll
    for (int w = 0; w < 8; ++w) {
      S0 += red[w][0]; Q0 += red[w][1]; S1 += red[w][2];
      Q1 += red[w][3]; S2 += red[w][4]; Q2 += red[w][5];
    }
    float mu0 = S0 * (1.f / CTAB), mu1 = S1 * (1.f / CTAB), mu2 = S2 * (1.f / CTAB);
    float r0 = rsqrtf(Q0 * (1.f / CTAB) - mu0 * mu0 + 1e-5f);
    float r1 = rsqrtf(Q1 * (1.f / CTAB) - mu1 * mu1 + 1e-5f);
    float r2 = rsqrtf(Q2 * (1.f / CTAB) - mu2 * mu2 + 1e-5f);

    float* orow = out + ((size_t)n * LSEQ + l) * OUTW;
    orow[c]            = (pm - mu0) * r0;
    orow[CTAB + c]     = (pa - mu1) * r1;
    orow[2 * CTAB + c] = (ps - mu2) * r2;
    if (c < DEMO) orow[3 * CTAB + c] = dval;
  }
}

// ---------------- launcher ----------------
extern "C" void kernel_launch(void* const* d_in, const int* in_sizes, int n_in,
                              void* d_out, int out_size, void* d_ws, size_t ws_size,
                              hipStream_t stream) {
  (void)in_sizes; (void)n_in; (void)out_size; (void)ws_size;
  const float* dem  = (const float*)d_in[0];
  const float* ts0  = (const float*)d_in[1];
  const float* ts1  = (const float*)d_in[2];
  const float* ts2  = (const float*)d_in[3];
  const float* ts3  = (const float*)d_in[4];
  const float* W1   = (const float*)d_in[5];
  const float* b1   = (const float*)d_in[6];
  const float* W2   = (const float*)d_in[7];
  const float* b2   = (const float*)d_in[8];
  const float* Wlin = (const float*)d_in[9];
  const float* blin = (const float*)d_in[10];

  // workspace layout
  float* wsf      = (float*)d_ws;
  float* dvec     = wsf;                                   // 640 (pad to 1024)
  float* xbuf     = wsf + 1024;                            // 65536*256 f32 (64 MB)
  float* aggmax   = xbuf + (size_t)NPTS * LSEQ * CTAB;     // 131072 each
  float* aggsum   = aggmax + NPTS * NCHUNK * CTAB;
  float* carrymax = aggsum + NPTS * NCHUNK * CTAB;
  float* carrysum = carrymax + NPTS * NCHUNK * CTAB;
  int*   aggarg   = (int*)(carrysum + NPTS * NCHUNK * CTAB);
  int*   carryarg = aggarg + NPTS * NCHUNK * CTAB;

  // output layout: out [N,L,788] f32 | time_inds [N,256,L] i32 | activations [N,256,L] f32
  float* out   = (float*)d_out;
  int*   tinds = (int*)((float*)d_out + (size_t)NPTS * LSEQ * OUTW);
  float* acts  = (float*)(tinds + (size_t)NPTS * CTAB * LSEQ);

  dem_mlp_kernel<<<1, 32, 0, stream>>>(dem, W1, b1, W2, b2, dvec);

  dim3 ggrid(16, 512);  // x = N-tiles (fast) so blocks sharing A-rows are L2-adjacent
  gemm_kernel<<<ggrid, 256, 0, stream>>>(ts0, ts1, ts2, ts3, dvec, Wlin, blin, xbuf);

  scan_agg_kernel<<<NPTS * NCHUNK, 256, 0, stream>>>(xbuf, aggmax, aggarg, aggsum);
  scan_carry_kernel<<<NPTS, 256, 0, stream>>>(aggmax, aggarg, aggsum,
                                              carrymax, carryarg, carrysum);
  scan_final_kernel<<<NPTS * NCHUNK, 256, 0, stream>>>(xbuf, carrymax, carryarg, carrysum,
                                                       dvec, out, tinds, acts);
}
